// NonLocalBlock_25331717112133
// MI455X (gfx1250) — matrix operand order repaired and checked by reading, not compiled, not run on hardware
//
#include <hip/hip_runtime.h>

typedef __attribute__((ext_vector_type(16))) _Float16 v16h;
typedef __attribute__((ext_vector_type(8)))  _Float16 v8h;
typedef __attribute__((ext_vector_type(8)))  float    v8f;

union FragU { v16h v; v8h h[2]; };

#define LDK 40           // padded K stride (halves) for LDS tiles
#define LOG2E 1.44269504088896340736f

// ---- CDNA5 Tensor Data Mover support (guarded; fallback = cooperative copy) ----
#if defined(__has_builtin)
#if __has_builtin(__builtin_amdgcn_tensor_load_to_lds) && \
    __has_builtin(__builtin_amdgcn_s_wait_tensorcnt)
#define USE_TDM 1
#endif
#endif
#ifndef USE_TDM
#define USE_TDM 0
#endif

#if USE_TDM
typedef __attribute__((ext_vector_type(4))) unsigned int u32x4;
typedef __attribute__((ext_vector_type(8))) int i32x8;
typedef __attribute__((ext_vector_type(4))) int i32x4;

__device__ __forceinline__ uint32_t lds_off32(const void* p) {
  // flat LDS address: aperture in bits 63:32, workgroup LDS offset in bits 31:0
  return (uint32_t)(uintptr_t)p;
}

// 2D tile DMA: tile_h rows of tile_w f16 elements, row stride stride_elems,
// LDS rows padded by 4 DWORDs every (2<<pad_iv) DWORDs (pad_iv: 3 -> 16DW=32h rows,
// 5 -> 64DW=128h rows), i.e. LDS row pitch = tile_w + 8 halves.
__device__ __forceinline__ void tdm2d(const void* gptr, uint32_t lds,
                                      uint32_t tile_w, uint32_t tile_h,
                                      uint32_t stride_elems, uint32_t pad_iv) {
  uint64_t ga = (uint64_t)(uintptr_t)gptr;
  u32x4 g0;
  g0[0] = 1u;                                              // count=1, user desc
  g0[1] = lds;                                             // lds_addr
  g0[2] = (uint32_t)ga;                                    // global_addr[31:0]
  g0[3] = (uint32_t)((ga >> 32) & 0x01ffffffu) | 0x80000000u;  // addr[56:32] | type=2
  const uint32_t td0 = 1u << 30, td1 = 1u << 30;           // huge tensor dims (never OOB)
  i32x8 g1;
  g1[0] = (int)((1u << 16) |               // data_size = 1 (2 bytes)
                (1u << 20) |               // pad_enable
                (pad_iv << 22) |           // pad_interval code
                (3u << 25));              // pad_amount code: 4 DWORDs = 8 halves
  g1[1] = (int)((td0 & 0xffffu) << 16);                       // tensor_dim0[15:0]
  g1[2] = (int)((td0 >> 16) | ((td1 & 0xffffu) << 16));       // td0[31:16] | td1[15:0]
  g1[3] = (int)((td1 >> 16) | (tile_w << 16));                // td1[31:16] | tile_dim0
  g1[4] = (int)(tile_h & 0xffffu);                            // tile_dim1 (tile_dim2=0)
  g1[5] = (int)stride_elems;                                  // tensor_dim0_stride[31:0]
  g1[6] = 0;                                                  // stride hi | dim1_stride lo
  g1[7] = 0;
  i32x4 z4 = {0, 0, 0, 0};
  i32x8 z8 = {0, 0, 0, 0, 0, 0, 0, 0};
  // 6-arg form (amdgpu-toolchain/clang-23): (g0, g1, g2, g3, extra, cpol)
  __builtin_amdgcn_tensor_load_to_lds(g0, g1, z4, z4, z8, 0);
}
#endif

// ---------------- WMMA helpers (CDNA5 wave32 layouts) ----------------

__device__ __forceinline__ v8f wmma32(v16h a, v16h b, v8f c) {
  return __builtin_amdgcn_wmma_f32_16x16x32_f16(false, a, false, b, (short)0, c,
                                                false, false);
}

// A fragment 16x32: lanes 0-15: row=lane, K {0..7},{16..23}; lanes 16-31: K {8..15},{24..31}
__device__ __forceinline__ v16h frag_a(const _Float16* tile, int ldk, int k0) {
  int lane = threadIdx.x & 31;
  int row  = lane & 15;
  int lo   = (lane < 16) ? 0 : 8;
  FragU f;
  f.h[0] = *(const v8h*)(tile + (size_t)row * ldk + k0 + lo);
  f.h[1] = *(const v8h*)(tile + (size_t)row * ldk + k0 + 16 + lo);
  return f.v;
}

// B fragment 32x16 from B_t layout tile[n][k]: lanes 0-15: n=lane, K {0..7},{16..23}; 16-31: K {8..15},{24..31}
__device__ __forceinline__ v16h frag_b(const _Float16* tile, int ldk, int k0) {
  int lane = threadIdx.x & 31;
  int row  = lane & 15;
  int off  = (lane < 16) ? 0 : 8;
  FragU f;
  f.h[0] = *(const v8h*)(tile + (size_t)row * ldk + k0 + off);
  f.h[1] = *(const v8h*)(tile + (size_t)row * ldk + k0 + off + 16);
  return f.v;
}

// ---------------- packing / layout kernels ----------------

__global__ void pack_conv_w(const float* __restrict__ wsrc, _Float16* __restrict__ wdst) {
  size_t idx = (size_t)blockIdx.x * 256 + threadIdx.x;
  if (idx >= (size_t)9 * 256 * 256) return;
  int t = (int)(idx / (256 * 256));
  int rem = (int)(idx % (256 * 256));
  int o = rem >> 8, c = rem & 255;
  wdst[idx] = (_Float16)wsrc[((size_t)o * 256 + c) * 9 + t];
}

__global__ void f32_to_f16(const float* __restrict__ src, _Float16* __restrict__ dst, int n) {
  int idx = blockIdx.x * 256 + threadIdx.x;
  if (idx < n) dst[idx] = (_Float16)src[idx];
}

__global__ void transpose_x(const float* __restrict__ x, _Float16* __restrict__ xT) {
  size_t idx = (size_t)blockIdx.x * 256 + threadIdx.x;
  int n = (int)(idx & 4095);
  size_t bc = idx >> 12;
  int c = (int)(bc & 255);
  int b = (int)(bc >> 8);
  xT[((size_t)(b * 4096) + n) * 256 + c] = (_Float16)x[idx];
}

// ---------------- 3x3 conv as implicit GEMM (WMMA, double-buffered, 64x128 tile) ----------------
__global__ __launch_bounds__(256) void conv3x3_gemm(
    const _Float16* __restrict__ xT, const _Float16* __restrict__ wpack,
    const float* __restrict__ bias, float* __restrict__ out) {
  __shared__ _Float16 As[2][64 * LDK];
  __shared__ _Float16 Bs[2][128 * LDK];
  int m0 = blockIdx.x * 64;
  int o0 = blockIdx.y * 128;
  int b  = m0 >> 12;
  int tid = threadIdx.x;
  int wave = tid >> 5, mt = wave >> 1, np = (wave & 1) * 4;
  v8f acc[4] = {};
  int rA = tid >> 2, cA = (tid & 3) * 8;
  int n_img = (m0 & 4095) + rA;
  int h = n_img >> 6, w = n_img & 63;

  auto fill = [&](int buf, int s) {
    int t = s >> 3, kk = (s & 7) << 5;
    int dr = t / 3 - 1, dc = t % 3 - 1;
    bool valid = ((unsigned)(h + dr) < 64u) && ((unsigned)(w + dc) < 64u);
    v8h z = {};
    const _Float16* src =
        xT + ((size_t)(b * 4096) + (n_img + dr * 64 + dc)) * 256 + kk + cA;
    *(v8h*)(&As[buf][rA * LDK + cA]) = valid ? *(const v8h*)src : z;
#pragma unroll
    for (int j = 0; j < 2; ++j) {
      int idx = tid + j * 256;
      int rB = idx >> 2, cB = (idx & 3) * 8;
      *(v8h*)(&Bs[buf][rB * LDK + cB]) =
          *(const v8h*)(wpack + ((size_t)t * 256 + o0 + rB) * 256 + kk + cB);
    }
  };

  fill(0, 0);
  for (int s = 0; s < 72; ++s) {
    int cur = s & 1, nxt = cur ^ 1;
    if (s + 1 < 72) fill(nxt, s + 1);
    __syncthreads();
    v16h a = frag_a(&As[cur][mt * 16 * LDK], LDK, 0);
#pragma unroll
    for (int j = 0; j < 4; ++j)
      acc[j] = wmma32(a, frag_b(&Bs[cur][((np + j) * 16) * LDK], LDK, 0), acc[j]);
    __syncthreads();
  }
  int lane = tid & 31;
#pragma unroll
  for (int j = 0; j < 4; ++j) {
    int nl = o0 + (np + j) * 16 + (lane & 15);
    float bv = bias[nl];
    for (int v = 0; v < 8; ++v) {
      int m = m0 + mt * 16 + v + ((lane < 16) ? 0 : 8);
      out[(size_t)m * 256 + nl] = acc[j][v] + bv;
    }
  }
}

// ---------------- generic 1x1 GEMM (WMMA, TDM double-buffered, 64x128 tile) ----------------
// out(m,n) = sum_k A[m][k]*Bw[n][k] + bias[n]
// mode 0: f32 [M][Nout]; mode 1: f16 [M][Nout]; mode 2: f16 transposed [B][Nout][4096]
__global__ __launch_bounds__(256) void gemm_f16(
    const _Float16* __restrict__ A, const _Float16* __restrict__ Bw,
    const float* __restrict__ bias, void* __restrict__ out,
    int K, int Nout, int mode) {
  __shared__ _Float16 As[2][64 * LDK];
  __shared__ _Float16 Bs[2][128 * LDK];
  int m0 = blockIdx.x * 64;
  int n0 = blockIdx.y * 128;
  int tid = threadIdx.x;
  int wave = tid >> 5, mt = wave >> 1, np = (wave & 1) * 4;
  v8f acc[4] = {};
  int steps = K >> 5;
#if USE_TDM
  bool w0 = (tid < 32);
  if (w0) {
    tdm2d(A  + (size_t)m0 * K, lds_off32(&As[0][0]), 32, 64,  (uint32_t)K, 3);
    tdm2d(Bw + (size_t)n0 * K, lds_off32(&Bs[0][0]), 32, 128, (uint32_t)K, 3);
  }
#else
  int rA = tid >> 2, cA = (tid & 3) * 8;
  *(v8h*)(&As[0][rA * LDK + cA]) = *(const v8h*)(A + (size_t)(m0 + rA) * K + cA);
  for (int j = 0; j < 2; ++j) {
    int idx = tid + j * 256;
    int rB = idx >> 2, cB = (idx & 3) * 8;
    *(v8h*)(&Bs[0][rB * LDK + cB]) = *(const v8h*)(Bw + (size_t)(n0 + rB) * K + cB);
  }
#endif
  for (int i = 0; i < steps; ++i) {
    int cur = i & 1, nxt = cur ^ 1;
    int k1 = (i + 1) << 5;
#if USE_TDM
    if (w0) {
      if (i + 1 < steps) {
        tdm2d(A  + (size_t)m0 * K + k1, lds_off32(&As[nxt][0]), 32, 64,  (uint32_t)K, 3);
        tdm2d(Bw + (size_t)n0 * K + k1, lds_off32(&Bs[nxt][0]), 32, 128, (uint32_t)K, 3);
        __builtin_amdgcn_s_wait_tensorcnt(2);   // current tile's 2 loads done (in-order)
      } else {
        __builtin_amdgcn_s_wait_tensorcnt(0);
      }
    }
#else
    if (i + 1 < steps) {
      *(v8h*)(&As[nxt][rA * LDK + cA]) = *(const v8h*)(A + (size_t)(m0 + rA) * K + k1 + cA);
      for (int j = 0; j < 2; ++j) {
        int idx = tid + j * 256;
        int rB = idx >> 2, cB = (idx & 3) * 8;
        *(v8h*)(&Bs[nxt][rB * LDK + cB]) =
            *(const v8h*)(Bw + (size_t)(n0 + rB) * K + k1 + cB);
      }
    }
#endif
    __syncthreads();
    v16h a = frag_a(&As[cur][mt * 16 * LDK], LDK, 0);
#pragma unroll
    for (int j = 0; j < 4; ++j)
      acc[j] = wmma32(a, frag_b(&Bs[cur][((np + j) * 16) * LDK], LDK, 0), acc[j]);
    __syncthreads();
  }
  int lane = tid & 31;
#pragma unroll
  for (int j = 0; j < 4; ++j) {
    int nl = n0 + (np + j) * 16 + (lane & 15);
    float bv = bias ? bias[nl] : 0.f;
    for (int v = 0; v < 8; ++v) {
      int m = m0 + mt * 16 + v + ((lane < 16) ? 0 : 8);
      float xv = acc[j][v] + bv;
      if (mode == 0) {
        ((float*)out)[(size_t)m * Nout + nl] = xv;
      } else if (mode == 1) {
        ((_Float16*)out)[(size_t)m * Nout + nl] = (_Float16)xv;
      } else {
        int bb = m >> 12, n = m & 4095;
        ((_Float16*)out)[((size_t)(bb * Nout + nl) << 12) + n] = (_Float16)xv;
      }
    }
  }
}

// ---------------- GroupNorm stats / apply ----------------

__global__ __launch_bounds__(256) void gn_stats(const float* __restrict__ xin,
                                                float* __restrict__ stats) {
  const int Nn = 4096, C = 256;
  int bg = blockIdx.x;
  int b = bg >> 3, g = bg & 7;
  int tid = threadIdx.x;
  float s = 0.f, s2 = 0.f;
  const float* base = xin + (size_t)b * Nn * C + g * 32;
  for (int i = tid; i < Nn * 32; i += 256) {
    int n = i >> 5, c = i & 31;
    float v = base[(size_t)n * C + c];
    s += v; s2 += v * v;
  }
  __shared__ float sh[256], sh2[256];
  sh[tid] = s; sh2[tid] = s2; __syncthreads();
  for (int d = 128; d > 0; d >>= 1) {
    if (tid < d) { sh[tid] += sh[tid + d]; sh2[tid] += sh2[tid + d]; }
    __syncthreads();
  }
  if (tid == 0) {
    float inv = 1.0f / (Nn * 32);
    float mean = sh[0] * inv;
    float var = sh2[0] * inv - mean * mean;
    stats[bg * 2] = mean;
    stats[bg * 2 + 1] = rsqrtf(var + 1e-5f);
  }
}

__global__ void gn1_apply(const float* __restrict__ xin, const float* __restrict__ stats,
                          const float* __restrict__ gw, const float* __restrict__ gb,
                          float* __restrict__ xs_f, _Float16* __restrict__ xs_h) {
  size_t idx = (size_t)blockIdx.x * 256 + threadIdx.x;
  int c = (int)(idx & 255);
  int b = (int)(idx >> 20);
  int g = c >> 5;
  float mean = stats[(b * 8 + g) * 2], rstd = stats[(b * 8 + g) * 2 + 1];
  float v = (xin[idx] - mean) * rstd * gw[c] + gb[c];
  v = fmaxf(v, 0.f);
  xs_f[idx] = v;
  xs_h[idx] = (_Float16)v;
}

// ---------------- flash attention (online softmax, WMMA, TDM double-buffered) ----------------
// theta/phi: [B][N][Ci] f16 ; gT: [B][Ci][N] f16 ; y: [B][N][Ci] f16
__global__ __launch_bounds__(128) void flash_attn(
    const _Float16* __restrict__ theta, const _Float16* __restrict__ phi,
    const _Float16* __restrict__ gT, _Float16* __restrict__ y) {
  const int Ci = 128, N = 4096;
  __shared__ _Float16 Ks[2][32 * 136];
  __shared__ _Float16 Gs[2][128 * 40];
  __shared__ _Float16 Ps[4][16 * 40];
  int b = blockIdx.y;
  int q0 = blockIdx.x * 64;
  int tid = threadIdx.x, wave = tid >> 5, lane = tid & 31;
  int qrow = q0 + wave * 16;

  const _Float16* phiB = phi + (size_t)(b * N) * Ci;
  const _Float16* gB   = gT  + (size_t)(b * Ci) * N;

  v16h qa[4];
  const _Float16* qbase = theta + ((size_t)(b * N) + qrow) * Ci;
  for (int j = 0; j < 4; ++j) qa[j] = frag_a(qbase, Ci, j * 32);

  v8f acc[8];
  for (int t = 0; t < 8; ++t) acc[t] = (v8f){};
  float mrun[8], lrun[8];
  for (int v = 0; v < 8; ++v) { mrun[v] = -1e30f; lrun[v] = 0.f; }

#if USE_TDM
  bool w0 = (tid < 32);
  if (w0) {
    tdm2d(phiB, lds_off32(&Ks[0][0]), 128, 32, 128, 5);
    tdm2d(gB,   lds_off32(&Gs[0][0]),  32, 128, 4096, 3);
  }
#else
  for (int i2 = tid; i2 < 512; i2 += 128) {
    int key = i2 >> 4, cc = (i2 & 15) * 8;
    *(v8h*)(&Ks[0][key * 136 + cc]) = *(const v8h*)(phiB + (size_t)key * Ci + cc);
  }
  for (int i2 = tid; i2 < 512; i2 += 128) {
    int ci = i2 >> 2, kk2 = (i2 & 3) * 8;
    *(v8h*)(&Gs[0][ci * 40 + kk2]) = *(const v8h*)(gB + (size_t)ci * N + kk2);
  }
#endif

  for (int it = 0; it < 128; ++it) {
    int cur = it & 1, nxt = cur ^ 1;
    int kb1 = (it + 1) * 32;
#if USE_TDM
    if (w0) {
      if (it + 1 < 128) {
        tdm2d(phiB + (size_t)kb1 * Ci, lds_off32(&Ks[nxt][0]), 128, 32, 128, 5);
        tdm2d(gB + kb1,                lds_off32(&Gs[nxt][0]),  32, 128, 4096, 3);
        __builtin_amdgcn_s_wait_tensorcnt(2);
      } else {
        __builtin_amdgcn_s_wait_tensorcnt(0);
      }
    }
#else
    if (it + 1 < 128) {
      for (int i2 = tid; i2 < 512; i2 += 128) {
        int key = i2 >> 4, cc = (i2 & 15) * 8;
        *(v8h*)(&Ks[nxt][key * 136 + cc]) =
            *(const v8h*)(phiB + (size_t)(kb1 + key) * Ci + cc);
      }
      for (int i2 = tid; i2 < 512; i2 += 128) {
        int ci = i2 >> 2, kk2 = (i2 & 3) * 8;
        *(v8h*)(&Gs[nxt][ci * 40 + kk2]) =
            *(const v8h*)(gB + (size_t)ci * N + kb1 + kk2);
      }
    }
#endif
    __syncthreads();

    v8f s0 = {}, s1 = {};
    for (int j = 0; j < 4; ++j) {
      s0 = wmma32(qa[j], frag_b(&Ks[cur][0], 136, j * 32), s0);
      s1 = wmma32(qa[j], frag_b(&Ks[cur][16 * 136], 136, j * 32), s1);
    }

    float sc[8];
    for (int v = 0; v < 8; ++v) {
      float mx = fmaxf(s0[v], s1[v]);
      for (int d = 8; d >= 1; d >>= 1) mx = fmaxf(mx, __shfl_xor(mx, d, 32));
      float mnew = fmaxf(mrun[v], mx);
      float scale = exp2f((mrun[v] - mnew) * LOG2E);
      float p0 = exp2f((s0[v] - mnew) * LOG2E);
      float p1 = exp2f((s1[v] - mnew) * LOG2E);
      float rs = p0 + p1;
      for (int d = 8; d >= 1; d >>= 1) rs += __shfl_xor(rs, d, 32);
      lrun[v] = lrun[v] * scale + rs;
      mrun[v] = mnew;
      sc[v] = scale;
      int row = v + ((lane < 16) ? 0 : 8);
      int col = lane & 15;
      Ps[wave][row * 40 + col]      = (_Float16)p0;
      Ps[wave][row * 40 + col + 16] = (_Float16)p1;
    }
    for (int t = 0; t < 8; ++t)
      for (int v = 0; v < 8; ++v) acc[t][v] *= sc[v];

    // per-wave P round-trip through LDS (DS ops are in-order within a wave)
    v16h pa = frag_a(&Ps[wave][0], 40, 0);
    for (int t = 0; t < 8; ++t)
      acc[t] = wmma32(pa, frag_b(&Gs[cur][t * 16 * 40], 40, 0), acc[t]);
    __syncthreads();
  }

  for (int v = 0; v < 8; ++v) {
    int row = qrow + v + ((lane < 16) ? 0 : 8);
    float inv = 1.0f / lrun[v];
    for (int t = 0; t < 8; ++t) {
      int ci = t * 16 + (lane & 15);
      y[((size_t)(b * N) + row) * Ci + ci] = (_Float16)(acc[t][v] * inv);
    }
  }
}

// ---------------- fused GN2 + scale + residual + NCHW transpose ----------------

__global__ __launch_bounds__(256) void final_out(
    const float* __restrict__ wy, const float* __restrict__ stats,
    const float* __restrict__ gw, const float* __restrict__ gb,
    const float* __restrict__ xs_f, float* __restrict__ out) {
  __shared__ float tile[32][33];
  int n0 = blockIdx.x * 32, c0 = blockIdx.y * 32, b = blockIdx.z;
  int tid = threadIdx.x;
  int tc = tid & 31, tr = tid >> 5;
  for (int q = 0; q < 4; ++q) {
    int n = n0 + tr + q * 8;
    int c = c0 + tc;
    int g = c >> 5;
    float mean = stats[(b * 8 + g) * 2], rstd = stats[(b * 8 + g) * 2 + 1];
    size_t idx = ((size_t)(b * 4096) + n) * 256 + c;
    float v = (wy[idx] - mean) * rstd * gw[c] + gb[c];
    tile[tr + q * 8][tc] = v * 0.1f + xs_f[idx];
  }
  __syncthreads();
  for (int q = 0; q < 4; ++q) {
    int c = c0 + tr + q * 8;
    int n = n0 + tc;
    out[((size_t)(b * 256) + c) * 4096 + n] = tile[tc][tr + q * 8];
  }
}

// ---------------- launch ----------------

extern "C" void kernel_launch(void* const* d_in, const int* in_sizes, int n_in,
                              void* d_out, int out_size, void* d_ws, size_t ws_size,
                              hipStream_t stream) {
  (void)in_sizes; (void)n_in; (void)out_size; (void)ws_size;
  const float* x        = (const float*)d_in[0];
  const float* smooth_w = (const float*)d_in[1];
  const float* smooth_b = (const float*)d_in[2];
  const float* gn1_w    = (const float*)d_in[3];
  const float* gn1_b    = (const float*)d_in[4];
  const float* g_w      = (const float*)d_in[5];
  const float* g_b      = (const float*)d_in[6];
  const float* theta_w  = (const float*)d_in[7];
  const float* theta_b  = (const float*)d_in[8];
  const float* phi_w    = (const float*)d_in[9];
  const float* phi_b    = (const float*)d_in[10];
  const float* w_w      = (const float*)d_in[11];
  const float* w_b      = (const float*)d_in[12];
  const float* gn2_w    = (const float*)d_in[13];
  const float* gn2_b    = (const float*)d_in[14];
  float* out = (float*)d_out;

  uint8_t* ws = (uint8_t*)d_ws;
  size_t off = 0;
  auto take = [&](size_t bytes) -> void* {
    void* p = ws + off;
    off += (bytes + 255) & ~(size_t)255;
    return p;
  };
  const size_t BN = 16384;
  float*    convbuf = (float*)take(BN * 256 * 4);     // conv out, later wy
  float*    xs_f    = (float*)take(BN * 256 * 4);
  _Float16* xs_h    = (_Float16*)take(BN * 256 * 2);
  _Float16* xT      = (_Float16*)take(BN * 256 * 2);
  _Float16* theta   = (_Float16*)take(BN * 128 * 2);
  _Float16* phi     = (_Float16*)take(BN * 128 * 2);
  _Float16* gT      = (_Float16*)take(BN * 128 * 2);
  _Float16* yb      = (_Float16*)take(BN * 128 * 2);
  _Float16* wpack   = (_Float16*)take((size_t)9 * 256 * 256 * 2);
  _Float16* thw_h   = (_Float16*)take((size_t)128 * 256 * 2);
  _Float16* phw_h   = (_Float16*)take((size_t)128 * 256 * 2);
  _Float16* gw_h    = (_Float16*)take((size_t)128 * 256 * 2);
  _Float16* ww_h    = (_Float16*)take((size_t)256 * 128 * 2);
  float*    stats1  = (float*)take(64 * 4);
  float*    stats2  = (float*)take(64 * 4);

  pack_conv_w<<<2304, 256, 0, stream>>>(smooth_w, wpack);
  f32_to_f16<<<128, 256, 0, stream>>>(theta_w, thw_h, 128 * 256);
  f32_to_f16<<<128, 256, 0, stream>>>(phi_w,   phw_h, 128 * 256);
  f32_to_f16<<<128, 256, 0, stream>>>(g_w,     gw_h,  128 * 256);
  f32_to_f16<<<128, 256, 0, stream>>>(w_w,     ww_h,  256 * 128);
  transpose_x<<<16384, 256, 0, stream>>>(x, xT);

  conv3x3_gemm<<<dim3(256, 2), 256, 0, stream>>>(xT, wpack, smooth_b, convbuf);
  gn_stats<<<32, 256, 0, stream>>>(convbuf, stats1);
  gn1_apply<<<16384, 256, 0, stream>>>(convbuf, stats1, gn1_w, gn1_b, xs_f, xs_h);

  gemm_f16<<<dim3(256, 1), 256, 0, stream>>>(xs_h, thw_h, theta_b, theta, 256, 128, 1);
  gemm_f16<<<dim3(256, 1), 256, 0, stream>>>(xs_h, phw_h, phi_b,   phi,   256, 128, 1);
  gemm_f16<<<dim3(256, 1), 256, 0, stream>>>(xs_h, gw_h,  g_b,     gT,    256, 128, 2);

  flash_attn<<<dim3(64, 4), 128, 0, stream>>>(theta, phi, gT, yb);

  gemm_f16<<<dim3(256, 2), 256, 0, stream>>>(yb, ww_h, w_b, convbuf, 128, 256, 0);
  gn_stats<<<32, 256, 0, stream>>>(convbuf, stats2);
  final_out<<<dim3(128, 8, 4), 256, 0, stream>>>(convbuf, stats2, gn2_w, gn2_b, xs_f, out);
}
